// Conv2dAffine8bit_26285199852240
// MI455X (gfx1250) — matrix operand order repaired and checked by reading, not compiled, not run on hardware
//
#include <hip/hip_runtime.h>
#include <hip/hip_bf16.h>

typedef __attribute__((ext_vector_type(16))) _Float16 v16h;
typedef __attribute__((ext_vector_type(8)))  _Float16 v8h;
typedef __attribute__((ext_vector_type(8)))  float    v8f;

// pointer types matching __builtin_amdgcn_global_load_async_to_lds_b128
typedef int vsi4 __attribute__((vector_size(16)));
typedef __attribute__((address_space(1))) vsi4 g_vsi4;   // global
typedef __attribute__((address_space(3))) vsi4 l_vsi4;   // LDS

#define CIN   256
#define COUT  256
#define IMGW  56
#define HW    3136              // 56*56
#define PTOT  100352            // 32*3136
#define WELEM 589824            // 256*256*9
#define NSTAGE 36               // 9 (r,s) * 4 chunks of K=64

union V16 { v16h v; v8h h[2]; };

#if defined(__AMDGCN__) && __has_builtin(__builtin_amdgcn_global_load_async_to_lds_b128) && __has_builtin(__builtin_amdgcn_s_wait_asynccnt)
#define USE_ASYNC_LDS 1
#else
#define USE_ASYNC_LDS 0
#endif

// ---------------------------------------------------------------------------
// Kernel 1: per-tensor min/max -> scale, zero_point (single block)
// ---------------------------------------------------------------------------
__global__ __launch_bounds__(256)
void minmax_kernel(const float* __restrict__ w, float* __restrict__ sz) {
    __shared__ float smin[256];
    __shared__ float smax[256];
    float mn = 3.402823466e38f, mx = -3.402823466e38f;
    for (int i = threadIdx.x; i < WELEM; i += 256) {
        float v = w[i];
        mn = fminf(mn, v);
        mx = fmaxf(mx, v);
    }
    smin[threadIdx.x] = mn; smax[threadIdx.x] = mx;
    __syncthreads();
    for (int st = 128; st > 0; st >>= 1) {
        if ((int)threadIdx.x < st) {
            smin[threadIdx.x] = fminf(smin[threadIdx.x], smin[threadIdx.x + st]);
            smax[threadIdx.x] = fmaxf(smax[threadIdx.x], smax[threadIdx.x + st]);
        }
        __syncthreads();
    }
    if (threadIdx.x == 0) {
        float scale = (smax[0] - smin[0]) / 255.0f;
        float zp = rintf(fminf(fmaxf(255.0f - smax[0] / scale, 0.0f), 255.0f));
        sz[0] = scale;
        sz[1] = zp;
    }
}

// ---------------------------------------------------------------------------
// Kernel 2: quantize, keep integer part (exact in f16), repack OIHW -> [rs][k][c]
// ---------------------------------------------------------------------------
__global__ __launch_bounds__(256)
void pack_kernel(const float* __restrict__ w, const float* __restrict__ sz,
                 _Float16* __restrict__ wq) {
    int i = blockIdx.x * 256 + threadIdx.x;
    if (i >= WELEM) return;
    float scale = sz[0], zp = sz[1];
    float q  = rintf(fminf(fmaxf(zp + w[i] / scale, 0.0f), 255.0f));
    float ip = q - zp;                       // integer in [-255,255], exact in f16
    int k   = i / (CIN * 9);
    int rem = i - k * (CIN * 9);
    int c   = rem / 9;
    int rs  = rem - c * 9;
    wq[(size_t)rs * (COUT * CIN) + (size_t)k * CIN + c] = (_Float16)ip;
}

// ---------------------------------------------------------------------------
// Kernel 3: implicit-GEMM conv3x3 via V_WMMA_F32_16X16X32_F16
//   block tile M=128 x N=64, 8 waves (4x2), wave = 32x32
//   K staged 64 at a time, double-buffered LDS, async A-tile copies
// ---------------------------------------------------------------------------
__global__ __launch_bounds__(256)
void conv_wmma_kernel(const float* __restrict__ x,
                      const _Float16* __restrict__ wq,
                      const float* __restrict__ sz,
                      const float* __restrict__ bias,
                      float* __restrict__ out) {
    __shared__ __align__(16) _Float16 As[2][128 * 64];   // [m][k], k contiguous (64)
    __shared__ __align__(16) _Float16 Bs[2][64 * 72];    // [n][k], row padded to 72

    const int tid   = threadIdx.x;
    const int lane  = tid & 31;
    const int wave  = tid >> 5;
    const int mWave = (wave & 3) << 5;      // 0,32,64,96
    const int nWave = (wave >> 2) << 5;     // 0,32
    const int l16   = lane & 15;
    const int khalf = lane >> 4;

    const int blockM = blockIdx.x * 128;    // out-channel base
    const int blockN = blockIdx.y * 64;     // flattened spatial base

    // Per-thread B-load coordinates (n fixed per thread)
    const int bN   = tid & 63;
    const int cgrp = tid >> 6;              // 0..3
    const int p    = blockN + bN;
    const int img  = p / HW;
    const int hw   = p - img * HW;
    const int h    = hw / IMGW;
    const int w    = hw - h * IMGW;
    const float* xImg = x + (size_t)img * (CIN * HW);

    v8f acc[2][2] = {};
    float bReg[16];
#if !USE_ASYNC_LDS
    float4 aReg[4];
#endif

    // ---- A tile staging: 128 rows x 64 f16 per stage ----
    auto issueA = [&](int st, int bf) {
        const _Float16* base = wq + (size_t)(st >> 2) * (COUT * CIN) + ((st & 3) << 6);
#pragma unroll
        for (int t = 0; t < 4; ++t) {
            int chunk = tid + t * 256;              // 0..1023 chunks of 8 f16
            int row = chunk >> 3, part = chunk & 7;
            const _Float16* src = base + (size_t)(blockM + row) * CIN + part * 8;
            _Float16* dst = &As[bf][row * 64 + part * 8];
#if USE_ASYNC_LDS
            __builtin_amdgcn_global_load_async_to_lds_b128(
                (g_vsi4*)(void*)src, (l_vsi4*)dst, 0, 0);
#else
            aReg[t] = *(const float4*)src;
            (void)dst;
#endif
        }
    };
    auto commitA = [&](int bf) {
#if !USE_ASYNC_LDS
#pragma unroll
        for (int t = 0; t < 4; ++t) {
            int chunk = tid + t * 256;
            int row = chunk >> 3, part = chunk & 7;
            *(float4*)&As[bf][row * 64 + part * 8] = aReg[t];
        }
#endif
    };

    // ---- B tile staging: 64 n x 64 c per stage, boundary-masked f32->f16 ----
    auto issueB = [&](int st) {
        int rsI = st >> 2, c0 = (st & 3) << 6;
        int r = rsI / 3, s2 = rsI - 3 * r;
        int hh = h + r - 1, ww = w + s2 - 1;
        bool valid = (hh >= 0) & (hh < IMGW) & (ww >= 0) & (ww < IMGW);
        const float* xPos = xImg + hh * IMGW + ww;
#pragma unroll
        for (int ps = 0; ps < 8; ++ps) {
            int c = c0 + 2 * (cgrp + 4 * ps);
            bReg[2 * ps]     = valid ? xPos[(size_t)c * HW] : 0.0f;
            bReg[2 * ps + 1] = valid ? xPos[(size_t)(c + 1) * HW] : 0.0f;
        }
    };
    auto commitB = [&](int bf) {
#pragma unroll
        for (int ps = 0; ps < 8; ++ps) {
            int c = 2 * (cgrp + 4 * ps);
            union { _Float16 h2[2]; unsigned u; } pk;
            pk.h2[0] = (_Float16)bReg[2 * ps];
            pk.h2[1] = (_Float16)bReg[2 * ps + 1];
            *(unsigned*)&Bs[bf][bN * 72 + c] = pk.u;
        }
    };

    // ---- compute: two K=32 chunks -> 8 WMMA per wave per stage ----
    auto compute = [&](int bf) {
#pragma unroll
        for (int kc = 0; kc < 2; ++kc) {
            V16 aF[2], bF[2];
#pragma unroll
            for (int mi = 0; mi < 2; ++mi) {
                const _Float16* ap = &As[bf][(mWave + mi * 16 + l16) * 64 + kc * 32];
                aF[mi].h[0] = *(const v8h*)(ap + khalf * 8);        // K 0..7 / 8..15
                aF[mi].h[1] = *(const v8h*)(ap + 16 + khalf * 8);   // K 16..23 / 24..31
            }
#pragma unroll
            for (int ni = 0; ni < 2; ++ni) {
                const _Float16* bp = &Bs[bf][(nWave + ni * 16 + l16) * 72 + kc * 32 + khalf * 8];
                bF[ni].h[0] = *(const v8h*)bp;
                bF[ni].h[1] = *(const v8h*)(bp + 16);
            }
#pragma unroll
            for (int mi = 0; mi < 2; ++mi)
#pragma unroll
                for (int ni = 0; ni < 2; ++ni)
                    acc[mi][ni] = __builtin_amdgcn_wmma_f32_16x16x32_f16(
                        false, aF[mi].v, false, bF[ni].v,
                        (short)0, acc[mi][ni], false, false);
        }
    };

    // ---- pipelined main loop ----
    issueA(0, 0);
    issueB(0);
    commitA(0);
    commitB(0);
#if USE_ASYNC_LDS
    __builtin_amdgcn_s_wait_asynccnt(0);
#endif
    __syncthreads();

    int bf = 0;
    for (int st = 0; st < NSTAGE; ++st) {
        if (st + 1 < NSTAGE) {
            issueA(st + 1, bf ^ 1);     // async writes go straight to LDS
            issueB(st + 1);             // global reads land in registers
        }
        compute(bf);
        if (st + 1 < NSTAGE) {
            commitA(bf ^ 1);
            commitB(bf ^ 1);
#if USE_ASYNC_LDS
            __builtin_amdgcn_s_wait_asynccnt(0);
#endif
        }
        __syncthreads();
        bf ^= 1;
    }

    // ---- epilogue: out = scale*acc + bias, per 16x16 f32 C/D lane layout ----
    const float scale = sz[0];
#pragma unroll
    for (int mi = 0; mi < 2; ++mi) {
#pragma unroll
        for (int ni = 0; ni < 2; ++ni) {
            int p2   = blockN + nWave + ni * 16 + l16;   // N = lane%16
            int img2 = p2 / HW;
            int hw2  = p2 - img2 * HW;
            float* op = out + (size_t)img2 * (COUT * HW) + hw2;
            int kBase = blockM + mWave + mi * 16 + khalf * 8;  // M = 8*(lane/16)+j
#pragma unroll
            for (int j = 0; j < 8; ++j) {
                int k = kBase + j;
                op[(size_t)k * HW] = scale * acc[mi][ni][j] + bias[k];
            }
        }
    }
}

// ---------------------------------------------------------------------------
extern "C" void kernel_launch(void* const* d_in, const int* in_sizes, int n_in,
                              void* d_out, int out_size, void* d_ws, size_t ws_size,
                              hipStream_t stream) {
    const float* x      = (const float*)d_in[0];
    const float* weight = (const float*)d_in[1];
    const float* bias   = (const float*)d_in[2];
    float* out = (float*)d_out;

    float*    sz = (float*)d_ws;                          // [scale, zero_point]
    _Float16* wq = (_Float16*)((char*)d_ws + 256);        // 1.18 MB packed weight

    minmax_kernel<<<1, 256, 0, stream>>>(weight, sz);
    pack_kernel<<<(WELEM + 255) / 256, 256, 0, stream>>>(weight, sz, wq);

    dim3 grid(COUT / 128, PTOT / 64);                     // 2 x 1568
    conv_wmma_kernel<<<grid, 256, 0, stream>>>(x, wq, sz, bias, out);
}
